// FeatureAlign_V2_44066364457482
// MI455X (gfx1250) — matrix operand order repaired and checked by reading, not compiled, not run on hardware
//
#include <hip/hip_runtime.h>
#include <hip/hip_bf16.h>

typedef __attribute__((ext_vector_type(16))) _Float16 v16h;
typedef __attribute__((ext_vector_type(2)))  _Float16 v2h;
typedef __attribute__((ext_vector_type(8)))  float    v8f;

#define Bb   2
#define Cc   128
#define Hh   128
#define Ww   128
#define Gg   8
#define Kk   9
#define HW   (Hh*Ww)           // 16384
#define NPIX (Bb*HW)           // 32768
#define CO3  (3*Gg*Kk)         // 216
#define CO3P 224               // padded to 14*16

__device__ __forceinline__ v8f wmma16(v16h a, v16h b, v8f c) {
  // v_wmma_f32_16x16x32_f16 : D = A(16x32 f16) x B(32x16 f16) + C(16x16 f32)
  return __builtin_amdgcn_wmma_f32_16x16x32_f16(false, a, false, b, (short)0, c,
                                                false, false);
}

// A-fragment slot order within one 32-wide K block (contiguous, matches B):
//   half0 lanes: elems 0..15 = K0..15   -> slots 0..15
//   half1 lanes: elems 0..15 = K16..31  -> slots 16..31
__device__ __forceinline__ int a_slot(int r) {
  return r;
}

// ---------------- 0. pack weights -> f16 fragments ----------------
// src layout: w[o][k][tap]  (taps==1 for 1x1 convs)
// dst layout: wp[tap][o][k/32][32 slots]  (o zero-padded to Opad)
__global__ void pack_w_kernel(const float* __restrict__ w, _Float16* __restrict__ wp,
                              int O, int K, int taps, int Opad) {
  size_t total = (size_t)taps * Opad * K;
  for (size_t i = (size_t)blockIdx.x * 256 + threadIdx.x; i < total;
       i += (size_t)gridDim.x * 256) {
    int k = (int)(i % K);
    size_t t1 = i / K;
    int o = (int)(t1 % Opad);
    int tap = (int)(t1 / Opad);
    int kb = k >> 5, r = k & 31;
    float v = (o < O) ? w[((size_t)o * K + k) * taps + tap] : 0.f;
    wp[(((size_t)tap * Opad + o) * (K >> 5) + kb) * 32 + a_slot(r)] = (_Float16)v;
  }
}

// ---------------- 1. global average pool ----------------
__global__ void gap_kernel(const float* __restrict__ fl, float* __restrict__ gap) {
  int bc = blockIdx.x;
  const float* p = fl + (size_t)bc * HW;
  float s = 0.f;
  for (int i = threadIdx.x; i < HW; i += 256) s += p[i];
  __shared__ float red[256];
  red[threadIdx.x] = s; __syncthreads();
  for (int off = 128; off > 0; off >>= 1) {
    if (threadIdx.x < off) red[threadIdx.x] += red[threadIdx.x + off];
    __syncthreads();
  }
  if (threadIdx.x == 0) gap[bc] = red[0] * (1.f / (float)HW);
}

// ---------------- 2. gating vector t[b,c] = a*sigmoid(a) ----------------
__global__ void gate_kernel(const float* __restrict__ gap,
                            const float* __restrict__ w_atten,
                            float* __restrict__ tvec) {
  int i = threadIdx.x;
  int b = i >> 7, o = i & 127;
  float a = 0.f;
  for (int c = 0; c < Cc; ++c) a += w_atten[o * Cc + c] * gap[b * Cc + c];
  tvec[i] = a / (1.f + __expf(-a));
}

// ---------------- 3. bilinear 2x upsample (half-pixel) ----------------
__global__ void upsample_kernel(const float* __restrict__ fs, float* __restrict__ up) {
  size_t idx = (size_t)blockIdx.x * 256 + threadIdx.x;
  int x = idx & 127, y = (idx >> 7) & 127;
  int bc = (int)(idx >> 14);
  float sy = y * 0.5f - 0.25f, sx = x * 0.5f - 0.25f;
  float y0f = floorf(sy), x0f = floorf(sx);
  float wy = sy - y0f, wx = sx - x0f;
  int y0 = (int)y0f, x0 = (int)x0f;
  int y0c = min(max(y0, 0), 63), y1c = min(max(y0 + 1, 0), 63);
  int x0c = min(max(x0, 0), 63), x1c = min(max(x0 + 1, 0), 63);
  const float* p = fs + (size_t)bc * 64 * 64;
  float v00 = p[y0c * 64 + x0c], v01 = p[y0c * 64 + x1c];
  float v10 = p[y1c * 64 + x0c], v11 = p[y1c * 64 + x1c];
  up[idx] = v00 * (1 - wy) * (1 - wx) + v01 * (1 - wy) * wx +
            v10 * wy * (1 - wx) + v11 * wy * wx;
}

// ---------------- 4. feat_arm = w_sel @ (feat_l + t) ----------------
// B tile staged in LDS as f16, layout xs[n][k] (k contiguous -> v16h per lane)
__global__ void gemm_arm_kernel(const float* __restrict__ fl,
                                const float* __restrict__ tvec,
                                const _Float16* __restrict__ wp,
                                float* __restrict__ arm) {
  __shared__ __align__(32) _Float16 xs[16 * 32];
  int n0 = blockIdx.x * 16;
  int b = n0 / HW, p0 = n0 - b * HW;
  int tid = threadIdx.x, wave = tid >> 5, lane = tid & 31;
  int half = lane >> 4, col = lane & 15;
  int o0 = wave * 16;
  int nn = tid & 15, kp = tid >> 4;          // staging coords: 16 n x 16 k-pairs
  v8f acc = {};
  #pragma unroll
  for (int kb = 0; kb < 4; ++kb) {
    __syncthreads();
    {
      int c = kb * 32 + kp * 2;
      float va = fl[((size_t)b * Cc + c) * HW + p0 + nn] + tvec[b * Cc + c];
      float vb = fl[((size_t)b * Cc + c + 1) * HW + p0 + nn] + tvec[b * Cc + c + 1];
      *(v2h*)(xs + nn * 32 + kp * 2) = (v2h){(_Float16)va, (_Float16)vb};
    }
    __syncthreads();
    v16h a  = ((const v16h*)(wp + ((size_t)(o0 + col) * 4 + kb) * 32))[half];
    v16h bf = ((const v16h*)(xs + col * 32))[half];
    acc = wmma16(a, bf, acc);
  }
  #pragma unroll
  for (int r = 0; r < 8; ++r) {
    int o = o0 + r + 8 * half;
    arm[((size_t)b * Cc + o) * HW + p0 + col] = acc[r];
  }
}

// ---------------- 5. off_feat = w_off @ [arm ; 2*up] ----------------
__global__ void gemm_off_kernel(const float* __restrict__ arm,
                                const float* __restrict__ up,
                                const _Float16* __restrict__ wp,
                                float* __restrict__ offf) {
  __shared__ __align__(32) _Float16 xs[16 * 32];
  int n0 = blockIdx.x * 16;
  int b = n0 / HW, p0 = n0 - b * HW;
  int tid = threadIdx.x, wave = tid >> 5, lane = tid & 31;
  int half = lane >> 4, col = lane & 15;
  int o0 = wave * 16;
  int nn = tid & 15, kp = tid >> 4;
  v8f acc = {};
  #pragma unroll
  for (int kb = 0; kb < 8; ++kb) {
    __syncthreads();
    {
      int c = kb * 32 + kp * 2;
      float va, vb;
      if (c < Cc) {
        va = arm[((size_t)b * Cc + c) * HW + p0 + nn];
        vb = arm[((size_t)b * Cc + c + 1) * HW + p0 + nn];
      } else {
        va = 2.f * up[((size_t)b * Cc + (c - Cc)) * HW + p0 + nn];
        vb = 2.f * up[((size_t)b * Cc + (c - Cc) + 1) * HW + p0 + nn];
      }
      *(v2h*)(xs + nn * 32 + kp * 2) = (v2h){(_Float16)va, (_Float16)vb};
    }
    __syncthreads();
    v16h a  = ((const v16h*)(wp + ((size_t)(o0 + col) * 8 + kb) * 32))[half];
    v16h bf = ((const v16h*)(xs + col * 32))[half];
    acc = wmma16(a, bf, acc);
  }
  #pragma unroll
  for (int r = 0; r < 8; ++r) {
    int o = o0 + r + 8 * half;
    offf[((size_t)b * Cc + o) * HW + p0 + col] = acc[r];
  }
}

// ---------------- 6. 3x3 conv (offsets/mask), implicit WMMA GEMM ----------
// grid: (B*H rows, 14 o-tiles); block 256 = 8 waves, each a 16-px subtile
__global__ void conv_off_kernel(const float* __restrict__ offf,
                                const _Float16* __restrict__ wp,
                                const float* __restrict__ b_coff,
                                float* __restrict__ raw) {
  __shared__ __align__(32) _Float16 xs[128 * 32];     // [x][k] f16, 8KB
  int rowid = blockIdx.x;
  int b = rowid / Hh, y = rowid % Hh;
  int o0 = blockIdx.y * 16;
  int tid = threadIdx.x, wave = tid >> 5, lane = tid & 31;
  int half = lane >> 4, col = lane & 15;
  int n0 = wave * 16;
  v8f acc = {};
  for (int tap = 0; tap < 9; ++tap) {
    int dy = tap / 3 - 1, dx = tap % 3 - 1;
    int yy = y + dy;
    bool yok = (yy >= 0) && (yy < Hh);
    for (int kb = 0; kb < 4; ++kb) {
      __syncthreads();
      for (int r = 0, e = tid; r < 8; ++r, e += 256) {   // 2048 k-pairs
        int kp2 = e >> 7, x = e & 127;
        int c = kb * 32 + kp2 * 2;
        int xx = x + dx;
        float va = 0.f, vb = 0.f;
        if (yok && xx >= 0 && xx < Ww) {
          va = offf[((size_t)b * Cc + c) * HW + yy * Ww + xx];
          vb = offf[((size_t)b * Cc + c + 1) * HW + yy * Ww + xx];
        }
        *(v2h*)(xs + x * 32 + kp2 * 2) = (v2h){(_Float16)va, (_Float16)vb};
      }
      __syncthreads();
      const _Float16* wb =
          wp + (((size_t)tap * CO3P + (o0 + col)) * 4 + kb) * 32;
      __builtin_prefetch(wb + CO3P * 4 * 32, 0, 1);     // next tap's block
      v16h a  = ((const v16h*)wb)[half];
      v16h bf = ((const v16h*)(xs + (n0 + col) * 32))[half];
      acc = wmma16(a, bf, acc);
    }
  }
  #pragma unroll
  for (int r = 0; r < 8; ++r) {
    int o = o0 + r + 8 * half;
    if (o < CO3)
      raw[((size_t)b * CO3 + o) * HW + y * Ww + n0 + col] = acc[r] + b_coff[o];
  }
}

// ---------------- 7. modulated deformable conv + relu + residual ----------
// grid: B*H rows; block 256 = 8 waves (one 16-out-ch tile each, 8 n-subtiles)
__global__ void dcn_kernel(const float* __restrict__ up,
                           const float* __restrict__ raw,
                           const float* __restrict__ arm,
                           const _Float16* __restrict__ wp,
                           const float* __restrict__ b_dcn,
                           float* __restrict__ out) {
  __shared__ __align__(32) _Float16 xs[128 * 32];     // sampled*mask [x][k], 8KB
  __shared__ float offc[3 * Gg * Ww];                 // per tap: dy,dx,mask [g][x]
  int rowid = blockIdx.x;
  int b = rowid / Hh, y = rowid % Hh;
  int tid = threadIdx.x, wave = tid >> 5, lane = tid & 31;
  int half = lane >> 4, col = lane & 15;
  int o0 = wave * 16;
  v8f acc[8] = {};
  const float* rawb = raw + (size_t)b * CO3 * HW + (size_t)y * Ww;
  for (int tap = 0; tap < 9; ++tap) {
    float ky = (float)(tap / 3 - 1), kx = (float)(tap % 3 - 1);
    __syncthreads();
    for (int e = tid; e < 3 * Gg * Ww; e += 256) {
      int which = e / (Gg * Ww);                      // 0:dy 1:dx 2:mask
      int g = (e / Ww) & 7, x = e & 127;
      float v = rawb[(size_t)(which * Gg * Kk + g * Kk + tap) * HW + x];
      if (which == 2) v = 1.f / (1.f + __expf(-v));
      offc[e] = v;
    }
    __syncthreads();
    for (int kb = 0; kb < 4; ++kb) {
      __syncthreads();
      for (int r = 0, e = tid; r < 8; ++r, e += 256) { // 2048 sample pairs
        int kp2 = e >> 7, x = e & 127;
        int c = kb * 32 + kp2 * 2;                     // c,c+1 share a group
        int g = c >> 4;
        float dyv = offc[g * Ww + x];
        float dxv = offc[Gg * Ww + g * Ww + x];
        float m   = offc[2 * Gg * Ww + g * Ww + x];
        float py = (float)y + ky + dyv;
        float px = (float)x + kx + dxv;
        float y0f = floorf(py), x0f = floorf(px);
        int y0 = (int)y0f, x0 = (int)x0f;
        float wy = py - y0f, wx = px - x0f;
        float w00 = (1 - wy) * (1 - wx) * m, w01 = (1 - wy) * wx * m;
        float w10 = wy * (1 - wx) * m,       w11 = wy * wx * m;
        bool yi0 = (y0 >= 0) && (y0 < Hh), yi1 = (y0 + 1 >= 0) && (y0 + 1 < Hh);
        bool xi0 = (x0 >= 0) && (x0 < Ww), xi1 = (x0 + 1 >= 0) && (x0 + 1 < Ww);
        const float* pl0 = up + ((size_t)b * Cc + c) * HW;
        const float* pl1 = pl0 + HW;
        float sa = 0.f, sb = 0.f;
        if (yi0 && xi0) { int i = y0 * Ww + x0;           sa += w00 * pl0[i]; sb += w00 * pl1[i]; }
        if (yi0 && xi1) { int i = y0 * Ww + x0 + 1;       sa += w01 * pl0[i]; sb += w01 * pl1[i]; }
        if (yi1 && xi0) { int i = (y0 + 1) * Ww + x0;     sa += w10 * pl0[i]; sb += w10 * pl1[i]; }
        if (yi1 && xi1) { int i = (y0 + 1) * Ww + x0 + 1; sa += w11 * pl0[i]; sb += w11 * pl1[i]; }
        *(v2h*)(xs + x * 32 + kp2 * 2) = (v2h){(_Float16)sa, (_Float16)sb};
      }
      __syncthreads();
      const _Float16* wb = wp + (((size_t)tap * Cc + (o0 + col)) * 4 + kb) * 32;
      __builtin_prefetch(wb + Cc * 4 * 32, 0, 1);
      v16h a = ((const v16h*)wb)[half];
      #pragma unroll
      for (int nt = 0; nt < 8; ++nt) {
        v16h bf = ((const v16h*)(xs + (nt * 16 + col) * 32))[half];
        acc[nt] = wmma16(a, bf, acc[nt]);
      }
    }
  }
  #pragma unroll
  for (int nt = 0; nt < 8; ++nt) {
    #pragma unroll
    for (int r = 0; r < 8; ++r) {
      int o = o0 + r + 8 * half, x = nt * 16 + col;
      float v = acc[nt][r] + b_dcn[o];
      v = v > 0.f ? v : 0.f;
      size_t idx = ((size_t)b * Cc + o) * HW + (size_t)y * Ww + x;
      out[idx] = v + arm[idx];
    }
  }
}

extern "C" void kernel_launch(void* const* d_in, const int* in_sizes, int n_in,
                              void* d_out, int out_size, void* d_ws, size_t ws_size,
                              hipStream_t stream) {
  const float* feat_l  = (const float*)d_in[0];
  const float* feat_s  = (const float*)d_in[1];
  const float* w_atten = (const float*)d_in[2];
  const float* w_sel   = (const float*)d_in[3];
  const float* w_off   = (const float*)d_in[4];
  const float* w_coff  = (const float*)d_in[5];
  const float* b_coff  = (const float*)d_in[6];
  const float* w_dcn   = (const float*)d_in[7];
  const float* b_dcn   = (const float*)d_in[8];
  float* out = (float*)d_out;

  float* ws = (float*)d_ws;
  float* gap  = ws;                                   // 256
  float* tvec = gap + Bb * Cc;                        // 256
  float* up   = tvec + Bb * Cc;                       // B*C*H*W
  float* arm  = up + (size_t)Bb * Cc * HW;            // B*C*H*W
  float* offf = arm + (size_t)Bb * Cc * HW;           // B*C*H*W
  float* raw  = offf + (size_t)Bb * Cc * HW;          // B*216*H*W
  float* fend = raw + (size_t)Bb * CO3 * HW;

  _Float16* wp_sel  = (_Float16*)fend;                        // 128*4*32
  _Float16* wp_off  = wp_sel + (size_t)128 * 4 * 32;          // 128*8*32
  _Float16* wp_coff = wp_off + (size_t)128 * 8 * 32;          // 9*224*4*32
  _Float16* wp_dcn  = wp_coff + (size_t)9 * CO3P * 4 * 32;    // 9*128*4*32

  pack_w_kernel<<<64, 256, 0, stream>>>(w_sel,  wp_sel,  Cc, Cc,      1, Cc);
  pack_w_kernel<<<64, 256, 0, stream>>>(w_off,  wp_off,  Cc, 2 * Cc,  1, Cc);
  pack_w_kernel<<<256, 256, 0, stream>>>(w_coff, wp_coff, CO3, Cc,    9, CO3P);
  pack_w_kernel<<<256, 256, 0, stream>>>(w_dcn,  wp_dcn,  Cc, Cc,     9, Cc);

  gap_kernel<<<Bb * Cc, 256, 0, stream>>>(feat_l, gap);
  gate_kernel<<<1, 256, 0, stream>>>(gap, w_atten, tvec);
  upsample_kernel<<<(Bb * Cc * HW) / 256, 256, 0, stream>>>(feat_s, up);
  gemm_arm_kernel<<<NPIX / 16, 256, 0, stream>>>(feat_l, tvec, wp_sel, arm);
  gemm_off_kernel<<<NPIX / 16, 256, 0, stream>>>(arm, up, wp_off, offf);
  conv_off_kernel<<<dim3(Bb * Hh, 14), 256, 0, stream>>>(offf, wp_coff, b_coff, raw);
  dcn_kernel<<<Bb * Hh, 256, 0, stream>>>(up, raw, arm, wp_dcn, b_dcn, out);
}
